// SwinBlock_23802708754481
// MI455X (gfx1250) — matrix operand order repaired and checked by reading, not compiled, not run on hardware
//
#include <hip/hip_runtime.h>
#include <hip/hip_bf16.h>
#include <math.h>

#define DIM   384
#define HEADS 12
#define HD    32
#define WSZ   7
#define SSZ   3
#define NWIN  49            // valid tokens per window
#define NPAD  64            // padded tokens per window (4 WMMA tiles)
#define HGT   56
#define WID   56
#define BATCH 32
#define LTOK  (HGT*WID)     // 3136
#define NW    64            // windows per image (8x8)
#define BW    (BATCH*NW)    // 2048 windows
#define TOK   (BATCH*LTOK)  // 100352 tokens
#define MROWS (BW*NPAD)     // 131072 padded window rows
#define HID   (4*DIM)       // 1536
#define NEGF  (-1e9f)

typedef __attribute__((ext_vector_type(16))) __bf16 v16bf;
typedef __attribute__((ext_vector_type(8)))  float  v8f;

union AB16 { v16bf v; unsigned int u[8]; uint4 q[2]; };

enum { EP_QKV = 0, EP_GELU = 1, EP_RES = 2, EP_F32 = 3 };

__device__ __forceinline__ unsigned short f2bf(float f) {
  union { float f; unsigned u; } c; c.f = f;
  unsigned u = c.u;
  u += 0x7FFFu + ((u >> 16) & 1u);      // round-to-nearest-even
  return (unsigned short)(u >> 16);
}

// ---------------------------------------------------------------- utilities
// Pack fp32 weight W[K][N] into bf16 B-operand layout:
//   out[(((NT*KS + ks)*32) + L)*16 + h] = W[ks*32 + (L>>4)*16 + h][NT*16 + (L&15)]
// so a lane's 16 halfs are contiguous (two uint4 loads per operand).
__global__ __launch_bounds__(256)
void pack_w_bf16(const float* __restrict__ in, unsigned short* __restrict__ out,
                 int K, int N) {
  long i = (long)blockIdx.x * 256 + threadIdx.x;
  if (i >= (long)K * N) return;
  int h  = (int)(i & 15);
  int L  = (int)((i >> 4) & 31);
  long g = i >> 9;
  int KS = K >> 5;
  int ks = (int)(g % KS);
  int NT = (int)(g / KS);
  int n = NT * 16 + (L & 15);
  int k = ks * 32 + (L >> 4) * 16 + h;
  out[i] = f2bf(in[(size_t)k * N + n]);
}

__global__ __launch_bounds__(256) void zero_pad_rows(unsigned short* __restrict__ Xw) {
  long idx = (long)blockIdx.x * 256 + threadIdx.x;       // BW*15*DIM elements
  if (idx >= (long)BW * 15 * DIM) return;
  long winI = idx / (15 * DIM);
  long r    = idx % (15 * DIM);
  int  row  = NWIN + (int)(r / DIM);
  int  c    = (int)(r % DIM);
  Xw[((size_t)winI * NPAD + row) * DIM + c] = 0;
}

// LayerNorm over C=384; one wave per token. windowed=1: roll(-3,-3) + window
// partition into [BW, NPAD, DIM] bf16. windowed=0: plain [TOK, DIM] bf16.
__global__ __launch_bounds__(256)
void layernorm_k(const float* __restrict__ x, const float* __restrict__ w,
                 const float* __restrict__ b, unsigned short* __restrict__ out,
                 int windowed) {
  const int wave = threadIdx.x >> 5, lane = threadIdx.x & 31;
  const long t = (long)blockIdx.x * 8 + wave;
  const float* px = x + (size_t)t * DIM;
  float vals[12], s = 0.f, s2 = 0.f;
#pragma unroll
  for (int j = 0; j < 12; ++j) {
    float f = px[lane + 32 * j];
    vals[j] = f; s += f; s2 += f * f;
  }
#pragma unroll
  for (int m = 1; m < 32; m <<= 1) { s += __shfl_xor(s, m, 32); s2 += __shfl_xor(s2, m, 32); }
  const float mean = s * (1.0f / DIM);
  const float var  = s2 * (1.0f / DIM) - mean * mean;
  const float rstd = rsqrtf(var + 1e-5f);

  unsigned short* po;
  if (windowed) {
    int bI = (int)(t / LTOK), l = (int)(t % LTOK);
    int h = l / WID, wI = l % WID;
    int dh = (h + HGT - SSZ) % HGT, dw = (wI + WID - SSZ) % WID;   // roll -SS
    int winI = (dh / WSZ) * (WID / WSZ) + (dw / WSZ);
    int row  = (dh % WSZ) * WSZ + (dw % WSZ);
    po = out + ((size_t)(bI * NW + winI) * NPAD + row) * DIM;
  } else {
    po = out + (size_t)t * DIM;
  }
#pragma unroll
  for (int j = 0; j < 12; ++j) {
    int c = lane + 32 * j;
    po[c] = f2bf((vals[j] - mean) * rstd * w[c] + b[c]);
  }
}

// window reverse + roll(+3,+3) + residual:  X2 = x + unwindow(PO)
__global__ __launch_bounds__(256)
void unwindow_residual(const float* __restrict__ PO, const float* __restrict__ x,
                       float* __restrict__ X2) {
  long idx = (long)blockIdx.x * 256 + threadIdx.x;       // TOK*DIM
  if (idx >= (long)TOK * DIM) return;
  int  c = (int)(idx % DIM);
  long t = idx / DIM;
  int bI = (int)(t / LTOK), l = (int)(t % LTOK);
  int h = l / WID, wI = l % WID;
  int dh = (h + HGT - SSZ) % HGT, dw = (wI + WID - SSZ) % WID;
  int winI = (dh / WSZ) * (WID / WSZ) + (dw / WSZ);
  int row  = (dh % WSZ) * WSZ + (dw % WSZ);
  X2[idx] = x[idx] + PO[((size_t)(bI * NW + winI) * NPAD + row) * DIM + c];
}

// ---------------------------------------------------------------- WMMA GEMM
// C[M,N] = A[M,K](bf16) @ W(packed bf16) + bias, fused epilogue.
// No LDS: A operand = 2 contiguous uint4 per lane (k = 16*hi + s per lane, same order as the
// packed B); B operand = 2 contiguous uint4 from the packed weights.
// Block: 256 threads (8 waves); wave tile 32x64 -> block tile 256x64.
__global__ __launch_bounds__(256)
void gemm_bf16(const unsigned short* __restrict__ A,
               const unsigned short* __restrict__ Wp,   // packed layout
               const float* __restrict__ bias,
               const float* __restrict__ res,
               void* __restrict__ out,
               int M, int K, int N, int mode, float qscale) {
  const int tid  = threadIdx.x;
  const int wave = tid >> 5, lane = tid & 31;
  const int lm = lane & 15, hi = lane >> 4;
  const int n0 = blockIdx.x * 64;
  const int m0 = blockIdx.y * 256 + wave * 32;
  const int KS = K >> 5;

  v8f acc[2][4];
#pragma unroll
  for (int mtt = 0; mtt < 2; ++mtt)
#pragma unroll
    for (int nt = 0; nt < 4; ++nt)
#pragma unroll
      for (int e = 0; e < 8; ++e) acc[mtt][nt][e] = 0.0f;

  const unsigned short* pa0 = A + (size_t)(m0 + lm) * K + hi * 16;
  const unsigned short* pa1 = A + (size_t)(m0 + 16 + lm) * K + hi * 16;
  const unsigned short* pb  = Wp + ((size_t)(n0 >> 4) * KS) * 512 + lane * 16;

  for (int ks = 0; ks < KS; ++ks) {
    AB16 a0, a1;
    a0.q[0] = *(const uint4*)(pa0);
    a0.q[1] = *(const uint4*)(pa0 + 8);
    a1.q[0] = *(const uint4*)(pa1);
    a1.q[1] = *(const uint4*)(pa1 + 8);
    pa0 += 32; pa1 += 32;
#pragma unroll
    for (int nt = 0; nt < 4; ++nt) {
      AB16 b;
      const unsigned short* q = pb + ((size_t)nt * KS + ks) * 512;
      b.q[0] = *(const uint4*)q;
      b.q[1] = *(const uint4*)(q + 8);
      acc[0][nt] = __builtin_amdgcn_wmma_f32_16x16x32_bf16(
          false, a0.v, false, b.v, (short)0, acc[0][nt], false, false);
      acc[1][nt] = __builtin_amdgcn_wmma_f32_16x16x32_bf16(
          false, a1.v, false, b.v, (short)0, acc[1][nt], false, false);
    }
  }

#pragma unroll
  for (int mtt = 0; mtt < 2; ++mtt)
#pragma unroll
    for (int nt = 0; nt < 4; ++nt) {
      int col = n0 + nt * 16 + lm;
      float bb = bias[col];
#pragma unroll
      for (int v = 0; v < 8; ++v) {
        int row = m0 + mtt * 16 + v + hi * 8;
        size_t o = (size_t)row * N + col;
        float val = acc[mtt][nt][v] + bb;
        if (mode == EP_QKV) {
          if (col < DIM) val *= qscale;                  // fold 1/sqrt(hd) into Q
          ((unsigned short*)out)[o] = f2bf(val);
        } else if (mode == EP_GELU) {
          val = 0.5f * val * (1.0f + erff(val * 0.70710678118f));
          ((unsigned short*)out)[o] = f2bf(val);
        } else if (mode == EP_RES) {
          ((float*)out)[o] = val + res[o];
        } else {
          ((float*)out)[o] = val;
        }
      }
    }
}

// ---------------------------------------------------------------- attention
// One wave per (window, head). S = Q*Kt (scale pre-folded in Q), analytic
// shift mask, row softmax via shfl within 16-lane halves, P staged bf16 in
// LDS, V transposed once into LDS -> all P@V operands are b128 LDS loads.
__global__ __launch_bounds__(32)
void attn_k(const unsigned short* __restrict__ QKV, unsigned short* __restrict__ O) {
  __shared__ unsigned short Ps[NPAD][72];
  __shared__ unsigned short Vt[HD][72];
  const int lane = threadIdx.x;
  const int lm = lane & 15, hi = lane >> 4;
  const int win  = blockIdx.x / HEADS;
  const int head = blockIdx.x % HEADS;
  const int w64  = win & (NW - 1);
  const int wh = w64 >> 3, ww = w64 & 7;
  const size_t base = (size_t)win * NPAD * (3 * DIM);

  // K^T B-operands: per-lane halfs are kk = h + hi*16 -> one 32B chunk
  AB16 kb[4];
#pragma unroll
  for (int nt = 0; nt < 4; ++nt) {
    const unsigned short* p =
        QKV + base + (size_t)(nt * 16 + lm) * (3 * DIM) + DIM + head * HD + hi * 16;
    kb[nt].q[0] = ((const uint4*)p)[0];
    kb[nt].q[1] = ((const uint4*)p)[1];
  }

  // stage V transposed: Vt[d][j], packed u32 stores (j pairs adjacent)
  {
    union { uint4 q[2]; unsigned short s[32]; } r0, r1;
    const unsigned short* pv = QKV + base + (size_t)(2 * lane) * (3 * DIM) + 2 * DIM + head * HD;
    r0.q[0] = ((const uint4*)pv)[0];
    r0.q[1] = ((const uint4*)pv)[1];
    pv += 3 * DIM;
    r1.q[0] = ((const uint4*)pv)[0];
    r1.q[1] = ((const uint4*)pv)[1];
#pragma unroll
    for (int d = 0; d < HD; ++d)
      *(unsigned int*)&Vt[d][2 * lane] = (unsigned)r0.s[d] | ((unsigned)r1.s[d] << 16);
  }

  for (int mt = 0; mt < 4; ++mt) {
    AB16 qa;
    const unsigned short* p =
        QKV + base + (size_t)(mt * 16 + lm) * (3 * DIM) + head * HD + hi * 16;
    qa.q[0] = *(const uint4*)(p);
    qa.q[1] = *(const uint4*)(p + 8);
    v8f S[4];
#pragma unroll
    for (int nt = 0; nt < 4; ++nt) {
#pragma unroll
      for (int e = 0; e < 8; ++e) S[nt][e] = 0.0f;
      S[nt] = __builtin_amdgcn_wmma_f32_16x16x32_bf16(
          false, qa.v, false, kb[nt].v, (short)0, S[nt], false, false);
    }
    // shifted-window mask, analytic (H-ws=49, H-ss=53)
#pragma unroll
    for (int nt = 0; nt < 4; ++nt)
#pragma unroll
      for (int v = 0; v < 8; ++v) {
        int i = mt * 16 + v + hi * 8;
        int j = nt * 16 + lm;
        bool kill;
        if (j >= NWIN) kill = true;
        else if (i < NWIN) {
          int gh = wh * WSZ + i / WSZ, gw = ww * WSZ + i % WSZ;
          int jh = wh * WSZ + j / WSZ, jw = ww * WSZ + j % WSZ;
          int li = (gh < 49 ? 0 : (gh < 53 ? 1 : 2)) * 3 + (gw < 49 ? 0 : (gw < 53 ? 1 : 2));
          int lj = (jh < 49 ? 0 : (jh < 53 ? 1 : 2)) * 3 + (jw < 49 ? 0 : (jw < 53 ? 1 : 2));
          kill = (li != lj);
        } else kill = false;
        if (kill) S[nt][v] += NEGF;
      }
    // row softmax: a row lives on one 16-lane half across the 4 n-tiles
#pragma unroll
    for (int v = 0; v < 8; ++v) {
      float mx = S[0][v];
#pragma unroll
      for (int nt = 1; nt < 4; ++nt) mx = fmaxf(mx, S[nt][v]);
#pragma unroll
      for (int m = 1; m < 16; m <<= 1) mx = fmaxf(mx, __shfl_xor(mx, m, 32));
      float e[4], sum = 0.f;
#pragma unroll
      for (int nt = 0; nt < 4; ++nt) { e[nt] = __expf(S[nt][v] - mx); sum += e[nt]; }
#pragma unroll
      for (int m = 1; m < 16; m <<= 1) sum += __shfl_xor(sum, m, 32);
      float inv = 1.0f / sum;
      int i = mt * 16 + v + hi * 8;
#pragma unroll
      for (int nt = 0; nt < 4; ++nt) Ps[i][nt * 16 + lm] = f2bf(e[nt] * inv);
    }
  }
  __syncthreads();

  // V B-operands from Vt: contiguous 32B per lane -> 2 x ds_load_b128
  AB16 vb[2][2];
#pragma unroll
  for (int kt = 0; kt < 2; ++kt)
#pragma unroll
    for (int nt2 = 0; nt2 < 2; ++nt2) {
      const unsigned short* p = &Vt[nt2 * 16 + lm][kt * 32 + hi * 16];
      vb[kt][nt2].q[0] = ((const uint4*)p)[0];
      vb[kt][nt2].q[1] = ((const uint4*)p)[1];
    }

  for (int mt = 0; mt < 4; ++mt) {
    v8f Oc[2];
#pragma unroll
    for (int nt2 = 0; nt2 < 2; ++nt2)
#pragma unroll
      for (int e = 0; e < 8; ++e) Oc[nt2][e] = 0.0f;
#pragma unroll
    for (int kt = 0; kt < 2; ++kt) {
      AB16 pa;
      const unsigned short* p = &Ps[mt * 16 + lm][kt * 32 + hi * 16];
      pa.q[0] = *(const uint4*)(p);
      pa.q[1] = *(const uint4*)(p + 8);
#pragma unroll
      for (int nt2 = 0; nt2 < 2; ++nt2)
        Oc[nt2] = __builtin_amdgcn_wmma_f32_16x16x32_bf16(
            false, pa.v, false, vb[kt][nt2].v, (short)0, Oc[nt2], false, false);
    }
#pragma unroll
    for (int nt2 = 0; nt2 < 2; ++nt2)
#pragma unroll
      for (int v = 0; v < 8; ++v) {
        int row = mt * 16 + v + hi * 8;
        int col = head * HD + nt2 * 16 + lm;
        O[((size_t)win * NPAD + row) * DIM + col] = f2bf(Oc[nt2][v]);
      }
  }
}

// ---------------------------------------------------------------- launch
extern "C" void kernel_launch(void* const* d_in, const int* in_sizes, int n_in,
                              void* d_out, int out_size, void* d_ws, size_t ws_size,
                              hipStream_t stream) {
  const float* x      = (const float*)d_in[0];
  const float* n1w    = (const float*)d_in[1];
  const float* n1b    = (const float*)d_in[2];
  const float* qkv_w  = (const float*)d_in[3];
  const float* qkv_b  = (const float*)d_in[4];
  const float* proj_w = (const float*)d_in[5];
  const float* proj_b = (const float*)d_in[6];
  const float* n2w    = (const float*)d_in[7];
  const float* n2b    = (const float*)d_in[8];
  const float* fc1_w  = (const float*)d_in[9];
  const float* fc1_b  = (const float*)d_in[10];
  const float* fc2_w  = (const float*)d_in[11];
  const float* fc2_b  = (const float*)d_in[12];
  float* out = (float*)d_out;
  (void)in_sizes; (void)n_in; (void)out_size; (void)ws_size;

  char* ws = (char*)d_ws;
  size_t off = 0;
  auto alloc = [&](size_t bytes) {
    char* p = ws + off;
    off += (bytes + 255) & ~(size_t)255;
    return p;
  };
  unsigned short* Wq = (unsigned short*)alloc(2ull * DIM * 3 * DIM);
  unsigned short* Wp = (unsigned short*)alloc(2ull * DIM * DIM);
  unsigned short* W1 = (unsigned short*)alloc(2ull * DIM * HID);
  unsigned short* W2 = (unsigned short*)alloc(2ull * HID * DIM);
  // Region R1: Xw (bf16) -> attnOut (bf16) -> Xm (bf16)
  char* R1 = alloc(2ull * MROWS * DIM);
  // Region R2: QKV (bf16) -> projOut (f32) -> H1 (bf16)
  char* R2 = alloc(2ull * TOK * HID);
  float* X2 = (float*)alloc(4ull * TOK * DIM);

  unsigned short* Xw  = (unsigned short*)R1;
  unsigned short* AO  = (unsigned short*)R1;
  unsigned short* Xm  = (unsigned short*)R1;
  unsigned short* QKV = (unsigned short*)R2;
  float*          PO  = (float*)R2;
  unsigned short* H1  = (unsigned short*)R2;

  auto cdiv = [](long a, long b) { return (int)((a + b - 1) / b); };
  const float qscale = 0.17677669529663687f;   // 1/sqrt(32)

  // weight pack fp32 -> bf16 B-operand layout
  pack_w_bf16<<<cdiv((long)DIM * 3 * DIM, 256), 256, 0, stream>>>(qkv_w, Wq, DIM, 3 * DIM);
  pack_w_bf16<<<cdiv((long)DIM * DIM, 256), 256, 0, stream>>>(proj_w, Wp, DIM, DIM);
  pack_w_bf16<<<cdiv((long)DIM * HID, 256), 256, 0, stream>>>(fc1_w, W1, DIM, HID);
  pack_w_bf16<<<cdiv((long)HID * DIM, 256), 256, 0, stream>>>(fc2_w, W2, HID, DIM);

  // LN1 + roll + window partition (bf16), pad rows zeroed
  layernorm_k<<<TOK / 8, 256, 0, stream>>>(x, n1w, n1b, Xw, 1);
  zero_pad_rows<<<cdiv((long)BW * 15 * DIM, 256), 256, 0, stream>>>(Xw);

  // QKV projection (q pre-scaled)
  gemm_bf16<<<dim3((3 * DIM) / 64, MROWS / 256), 256, 0, stream>>>(
      Xw, Wq, qkv_b, nullptr, QKV, MROWS, DIM, 3 * DIM, EP_QKV, qscale);

  // windowed MHSA
  attn_k<<<BW * HEADS, 32, 0, stream>>>(QKV, AO);

  // output projection (fp32)
  gemm_bf16<<<dim3(DIM / 64, MROWS / 256), 256, 0, stream>>>(
      AO, Wp, proj_b, nullptr, PO, MROWS, DIM, DIM, EP_F32, 1.0f);

  // window reverse + roll back + residual
  unwindow_residual<<<cdiv((long)TOK * DIM, 256), 256, 0, stream>>>(PO, x, X2);

  // LN2 -> bf16
  layernorm_k<<<TOK / 8, 256, 0, stream>>>(X2, n2w, n2b, Xm, 0);

  // MLP: FC1 + GELU (bf16), FC2 + residual (fp32 -> d_out)
  gemm_bf16<<<dim3(HID / 64, TOK / 256), 256, 0, stream>>>(
      Xm, W1, fc1_b, nullptr, H1, TOK, DIM, HID, EP_GELU, 1.0f);
  gemm_bf16<<<dim3(DIM / 64, TOK / 256), 256, 0, stream>>>(
      H1, W2, fc2_b, X2, out, TOK, HID, DIM, EP_RES, 1.0f);
}
